// AutoEncoderTopK_9036611191359
// MI455X (gfx1250) — matrix operand order repaired and checked by reading, not compiled, not run on hardware
//
#include <hip/hip_runtime.h>
#include <cstdint>

#define B_DIM 8192
#define D_DIM 768
#define F_DIM 16384
#define K_TOP 32
#define GWAVES 4      // waves per encode block
#define WROWS  32     // rows per wave (2 x 16-row WMMA tiles)
#define F_SPLIT 2     // F-dimension split across gridDim.y

typedef __attribute__((ext_vector_type(16))) __bf16 v16bf;
typedef __attribute__((ext_vector_type(8)))  float  v8f;

// float -> bf16 (round to nearest even)
__device__ __forceinline__ uint16_t f2bf(float f) {
    union { float f; uint32_t u; } c; c.f = f;
    uint32_t u = c.u;
    uint32_t r = u + 0x7FFFu + ((u >> 16) & 1u);
    return (uint16_t)(r >> 16);
}

// ---------------- conversion kernels ----------------
__global__ void conv_x_kernel(const float* __restrict__ x,
                              const float* __restrict__ b_dec,
                              uint16_t* __restrict__ xb) {
    int id = blockIdx.x * 256 + threadIdx.x;           // B*D elements, exact grid
    xb[id] = f2bf(x[id] - b_dec[id % D_DIM]);
}

__global__ void conv_w_kernel(const float* __restrict__ w,
                              uint16_t* __restrict__ wb) {
    int id = blockIdx.x * 256 + threadIdx.x;           // F*D elements, exact grid
    wb[id] = f2bf(w[id]);
}

// W_dec[D,F] (row-major, d major) -> WdT[F,D]
__global__ void transpose_wdec_kernel(const float* __restrict__ Wdec,
                                      float* __restrict__ WdT) {
    __shared__ float tile[32][33];
    const int f0 = blockIdx.x * 32;
    const int d0 = blockIdx.y * 32;
    const int tx = threadIdx.x, ty = threadIdx.y;      // block (32, 8)
    #pragma unroll
    for (int j = 0; j < 4; ++j)
        tile[ty + 8 * j][tx] = Wdec[(size_t)(d0 + ty + 8 * j) * F_DIM + (f0 + tx)];
    __syncthreads();
    #pragma unroll
    for (int j = 0; j < 4; ++j)
        WdT[(size_t)(f0 + ty + 8 * j) * D_DIM + (d0 + tx)] = tile[tx][ty + 8 * j];
}

// ---------------- fused encoder GEMM + ReLU + partial top-K ----------------
// grid: (B/(GWAVES*WROWS), F_SPLIT) = (64, 2) blocks, 128 threads (4 waves).
// Each wave owns 32 rows (2 M-tiles); per F-step computes a 32x32 preact tile
// with 4 bf16 WMMA accumulators (2x2 register blocking) over K = D = 768.
// 8 x b128 loads feed 4 WMMAs -> 32 B/lane/WMMA.
// Each lane then owns one of the 32 rows for the top-K insertion scan.
__launch_bounds__(GWAVES * 32)
__global__ void encode_topk_kernel(const uint16_t* __restrict__ xb,
                                   const uint16_t* __restrict__ wb,
                                   const float* __restrict__ b_enc,
                                   float* __restrict__ pv_g,   // [B][F_SPLIT][K_TOP]
                                   int* __restrict__ pi_g) {
    __shared__ float stage[GWAVES][WROWS][33];
    __shared__ float topv[GWAVES][WROWS][33];
    __shared__ int   topi[GWAVES][WROWS][33];

    const int tid   = threadIdx.x;
    const int wave  = tid >> 5;
    const int lane  = tid & 31;
    const int n     = lane & 15;
    const int half  = lane >> 4;
    const int r0    = blockIdx.x * (GWAVES * WROWS) + wave * WROWS;
    const int fbase = blockIdx.y * (F_DIM / F_SPLIT);

    // lane l privately owns row r0+l : init its top-K list (LDS-resident)
    #pragma unroll
    for (int t = 0; t < K_TOP; ++t) { topv[wave][lane][t] = 0.f; topi[wave][lane][t] = 0; }
    float rmin = 0.f;
    int   rpos = 0;

    const uint16_t* arow0 = xb + (size_t)(r0 + n) * D_DIM;
    const uint16_t* arow1 = arow0 + (size_t)16 * D_DIM;
    const int ah = half * 8;    // A frag: K in {0..7,16..23} or {8..15,24..31}
    const int bh = half * 8;    // B frag: K in {0..7,16..23} or {8..15,24..31}

    for (int fo = 0; fo < F_DIM / F_SPLIT; fo += 32) {
        const int f0 = fbase + fo;
        const uint16_t* brow0 = wb + (size_t)(f0 + n) * D_DIM;
        const uint16_t* brow1 = brow0 + (size_t)16 * D_DIM;

        v8f acc00 = {}, acc01 = {}, acc10 = {}, acc11 = {};
        #pragma unroll 2
        for (int kt = 0; kt < D_DIM / 32; ++kt) {
            const int kb = kt * 32;
            union { v16bf v; uint4 q[2]; } a0, a1, b0, b1;
            a0.q[0] = *(const uint4*)(arow0 + kb + ah);
            a0.q[1] = *(const uint4*)(arow0 + kb + ah + 16);
            a1.q[0] = *(const uint4*)(arow1 + kb + ah);
            a1.q[1] = *(const uint4*)(arow1 + kb + ah + 16);
            b0.q[0] = *(const uint4*)(brow0 + kb + bh);
            b0.q[1] = *(const uint4*)(brow0 + kb + bh + 16);
            b1.q[0] = *(const uint4*)(brow1 + kb + bh);
            b1.q[1] = *(const uint4*)(brow1 + kb + bh + 16);
            acc00 = __builtin_amdgcn_wmma_f32_16x16x32_bf16(
                false, a0.v, false, b0.v, (short)0, acc00, false, false);
            acc01 = __builtin_amdgcn_wmma_f32_16x16x32_bf16(
                false, a0.v, false, b1.v, (short)0, acc01, false, false);
            acc10 = __builtin_amdgcn_wmma_f32_16x16x32_bf16(
                false, a1.v, false, b0.v, (short)0, acc10, false, false);
            acc11 = __builtin_amdgcn_wmma_f32_16x16x32_bf16(
                false, a1.v, false, b1.v, (short)0, acc11, false, false);
        }

        // bias + ReLU, stage 32x32 tile (C layout: lane n = col, VGPR v = row)
        const float bias0 = b_enc[f0 + n];
        const float bias1 = b_enc[f0 + 16 + n];
        const int mb = half * 8;
        #pragma unroll
        for (int v = 0; v < 8; ++v) {
            float p;
            p = acc00[v] + bias0; stage[wave][mb + v][n]           = p > 0.f ? p : 0.f;
            p = acc01[v] + bias1; stage[wave][mb + v][16 + n]      = p > 0.f ? p : 0.f;
            p = acc10[v] + bias0; stage[wave][16 + mb + v][n]      = p > 0.f ? p : 0.f;
            p = acc11[v] + bias1; stage[wave][16 + mb + v][16 + n] = p > 0.f ? p : 0.f;
        }
        __syncthreads();

        // per-row top-K insertion: lane l scans the 32 candidates of row l
        for (int j = 0; j < 32; ++j) {
            const float val = stage[wave][lane][j];
            if (val > rmin) {
                topv[wave][lane][rpos] = val;
                topi[wave][lane][rpos] = f0 + j;
                rmin = topv[wave][lane][0]; rpos = 0;
                #pragma unroll
                for (int t = 1; t < K_TOP; ++t) {
                    const float tvv = topv[wave][lane][t];
                    if (tvv < rmin) { rmin = tvv; rpos = t; }
                }
            }
        }
        __syncthreads();
    }

    // write this F-half's partial top-K list
    float* ov = pv_g + ((size_t)(r0 + lane) * F_SPLIT + blockIdx.y) * K_TOP;
    int*   oi = pi_g + ((size_t)(r0 + lane) * F_SPLIT + blockIdx.y) * K_TOP;
    #pragma unroll
    for (int t = 0; t < K_TOP; ++t) { ov[t] = topv[wave][lane][t]; oi[t] = topi[wave][lane][t]; }
}

// ---------------- merge partial top-K lists ----------------
// one thread per row: select top-32 of the F_SPLIT*32 = 64 candidates
__global__ void merge_topk_kernel(const float* __restrict__ pv,
                                  const int* __restrict__ pi,
                                  float* __restrict__ tv,
                                  int* __restrict__ ti) {
    const int row = blockIdx.x * 256 + threadIdx.x;    // B rows, exact grid
    const float* v  = pv + (size_t)row * (F_SPLIT * K_TOP);
    const int*   ix = pi + (size_t)row * (F_SPLIT * K_TOP);
    uint64_t taken = 0;
    for (int t = 0; t < K_TOP; ++t) {
        float best = -1.f; int bj = 0;
        for (int j = 0; j < F_SPLIT * K_TOP; ++j) {
            const float val = v[j];
            if (!((taken >> j) & 1) && val > best) { best = val; bj = j; }
        }
        taken |= (uint64_t)1 << bj;
        tv[(size_t)row * K_TOP + t] = best > 0.f ? best : 0.f;
        ti[(size_t)row * K_TOP + t] = best > 0.f ? ix[bj] : 0;
    }
}

// ---------------- sparse decoder ----------------
// one block per batch row: x_hat[b,:] = b_dec + sum_k val_k * WdT[idx_k, :]
__global__ void decode_kernel(const float* __restrict__ tv,
                              const int* __restrict__ ti,
                              const float* __restrict__ WdT,
                              const float* __restrict__ b_dec,
                              float* __restrict__ out) {
    __shared__ float sv[K_TOP];
    __shared__ int   si[K_TOP];
    const int b = blockIdx.x;
    if (threadIdx.x < K_TOP) {
        sv[threadIdx.x] = tv[(size_t)b * K_TOP + threadIdx.x];
        si[threadIdx.x] = ti[(size_t)b * K_TOP + threadIdx.x];
    }
    __syncthreads();
    for (int d = threadIdx.x; d < D_DIM; d += 256) {
        float acc = b_dec[d];
        #pragma unroll 8
        for (int k = 0; k < K_TOP; ++k)
            acc = fmaf(sv[k], WdT[(size_t)si[k] * D_DIM + d], acc);
        out[(size_t)b * D_DIM + d] = acc;
    }
}

// ---------------- launcher ----------------
extern "C" void kernel_launch(void* const* d_in, const int* in_sizes, int n_in,
                              void* d_out, int out_size, void* d_ws, size_t ws_size,
                              hipStream_t stream) {
    const float* x     = (const float*)d_in[0];
    const float* W_enc = (const float*)d_in[1];
    const float* b_enc = (const float*)d_in[2];
    const float* W_dec = (const float*)d_in[3];
    const float* b_dec = (const float*)d_in[4];
    float* out = (float*)d_out;
    (void)in_sizes; (void)n_in; (void)out_size; (void)ws_size;

    char* ws = (char*)d_ws;
    size_t off = 0;
    auto alloc = [&](size_t bytes) -> void* {
        void* p = ws + off;
        off += (bytes + 255) & ~(size_t)255;
        return p;
    };
    uint16_t* xb  = (uint16_t*)alloc((size_t)B_DIM * D_DIM * sizeof(uint16_t));
    uint16_t* wb  = (uint16_t*)alloc((size_t)F_DIM * D_DIM * sizeof(uint16_t));
    float*    WdT = (float*)   alloc((size_t)F_DIM * D_DIM * sizeof(float));
    float*    pv  = (float*)   alloc((size_t)B_DIM * F_SPLIT * K_TOP * sizeof(float));
    int*      pi  = (int*)     alloc((size_t)B_DIM * F_SPLIT * K_TOP * sizeof(int));
    float*    tv  = (float*)   alloc((size_t)B_DIM * K_TOP * sizeof(float));
    int*      ti  = (int*)     alloc((size_t)B_DIM * K_TOP * sizeof(int));

    conv_x_kernel<<<(B_DIM * D_DIM) / 256, 256, 0, stream>>>(x, b_dec, xb);
    conv_w_kernel<<<(F_DIM * D_DIM) / 256, 256, 0, stream>>>(W_enc, wb);
    transpose_wdec_kernel<<<dim3(F_DIM / 32, D_DIM / 32), dim3(32, 8), 0, stream>>>(W_dec, WdT);
    encode_topk_kernel<<<dim3(B_DIM / (GWAVES * WROWS), F_SPLIT), GWAVES * 32, 0, stream>>>(
        xb, wb, b_enc, pv, pi);
    merge_topk_kernel<<<B_DIM / 256, 256, 0, stream>>>(pv, pi, tv, ti);
    decode_kernel<<<B_DIM, 256, 0, stream>>>(tv, ti, WdT, b_dec, out);
}
